// TTF_RecAvg_44805098832420
// MI455X (gfx1250) — matrix operand order repaired and checked by reading, not compiled, not run on hardware
//
#include <hip/hip_runtime.h>

#define D_DIM  768
#define N_NOTES 512
#define T_DIM 1024
#define B_DIM   64

typedef __attribute__((ext_vector_type(16))) _Float16 v16h;
typedef __attribute__((ext_vector_type(8)))  _Float16 v8h;
typedef __attribute__((ext_vector_type(4)))  _Float16 v4h;
typedef __attribute__((ext_vector_type(8)))  float    v8f;

union U16h { v16h v; v8h h[2]; };

// ---------------- Kernel A: f32 -> f16 weight conversion --------------------
__global__ void __launch_bounds__(256) k_cvt_w(const float* __restrict__ Win,
                                               const float* __restrict__ Wp,
                                               _Float16* __restrict__ Win16,
                                               _Float16* __restrict__ Wp16) {
  int i = blockIdx.x * 256 + threadIdx.x;
  if (i < D_DIM * D_DIM) {
    Win16[i] = (_Float16)Win[i];
    Wp16[i]  = (_Float16)Wp[i];
  }
}

// ---------------- Kernel B: note_mask only (fallback path) ------------------
__global__ void __launch_bounds__(256) k_mask(const float* __restrict__ V,
                                              float* __restrict__ mask) {
  int wid  = threadIdx.x >> 5;
  int lane = threadIdx.x & 31;
  int row  = blockIdx.x * 8 + wid;              // 0 .. B*N-1
  const float* p = V + (size_t)row * D_DIM;
  float s = 0.f;
  for (int i = lane; i < D_DIM; i += 32) s += fabsf(p[i]);
  #pragma unroll
  for (int off = 16; off; off >>= 1) s += __shfl_xor(s, off, 32);
  if (lane == 0) mask[row] = (s > 0.f) ? 1.f : 0.f;
}

// ---------------- Kernel B': V -> f16 conversion fused with note_mask ------
// one wave per row; lane owns 4 consecutive floats of each 128-wide chunk
__global__ void __launch_bounds__(256) k_prep(const float* __restrict__ V,
                                              _Float16* __restrict__ Vh,
                                              float* __restrict__ mask) {
  int wid  = threadIdx.x >> 5;
  int lane = threadIdx.x & 31;
  int row  = blockIdx.x * 8 + wid;              // 0 .. B*N-1
  const float4* p  = (const float4*)(V + (size_t)row * D_DIM) + lane;
  v4h*          ph = (v4h*)(Vh + (size_t)row * D_DIM) + lane;
  float s = 0.f;
  #pragma unroll
  for (int i = 0; i < 6; i++) {                 // 6 * 128 = 768
    float4 f = p[i * 32];
    s += fabsf(f.x) + fabsf(f.y) + fabsf(f.z) + fabsf(f.w);
    v4h h; h[0] = (_Float16)f.x; h[1] = (_Float16)f.y;
           h[2] = (_Float16)f.z; h[3] = (_Float16)f.w;
    ph[i * 32] = h;
  }
  #pragma unroll
  for (int off = 16; off; off >>= 1) s += __shfl_xor(s, off, 32);
  if (lane == 0) mask[row] = (s > 0.f) ? 1.f : 0.f;
}

// ---------------- Kernel C: M_txt[b] = any(note_mask[b,:]) ------------------
__global__ void __launch_bounds__(256) k_mtxt(const float* __restrict__ mask,
                                              float* __restrict__ out_m) {
  __shared__ int flag;
  if (threadIdx.x == 0) flag = 0;
  __syncthreads();
  int b = blockIdx.x;
  float v = mask[b * N_NOTES + threadIdx.x] + mask[b * N_NOTES + 256 + threadIdx.x];
  if (v > 0.f) atomicOr(&flag, 1);
  __syncthreads();
  if (threadIdx.x == 0) out_m[b] = flag ? 1.f : 0.f;
}

// ---------------- shared GEMM1 epilogue -------------------------------------
__device__ __forceinline__ void gemm1_store(v8f* acc, const float* b_in,
                                            _Float16* Vpt, int m0, int e0,
                                            int lc, int half) {
  #pragma unroll
  for (int j = 0; j < 4; j++) {
    int e = e0 + j * 16 + lc;
    float bias = b_in[e];
    #pragma unroll
    for (int r = 0; r < 8; r++) {
      int m = m0 + r + (half ? 8 : 0);
      int b = m >> 9, n = m & (N_NOTES - 1);
      Vpt[(size_t)b * D_DIM * N_NOTES + (size_t)e * N_NOTES + n] = (_Float16)(acc[j][r] + bias);
    }
  }
}

// ---------------- Kernel D (f16-prepass path): GEMM1 from Vh ---------------
__global__ void __launch_bounds__(256) k_gemm1_f16(const _Float16* __restrict__ Vh,
                                                   const _Float16* __restrict__ W16,
                                                   const float* __restrict__ b_in,
                                                   _Float16* __restrict__ Vpt) {
  int wid  = threadIdx.x >> 5;
  int lane = threadIdx.x & 31;
  int gw   = blockIdx.x * 8 + wid;
  int mt   = gw / 12, ng = gw % 12;
  int m0   = mt * 16, e0 = ng * 64;
  int half = (lane < 16) ? 0 : 1;
  int lc   = lane & 15;

  v8f acc[4] = {};
  const _Float16* aptr = Vh + (size_t)(m0 + lc) * D_DIM + (half ? 8 : 0);

  for (int kk = 0; kk < D_DIM; kk += 32) {
    U16h A;
    A.h[0] = *(const v8h*)(aptr + kk);
    A.h[1] = *(const v8h*)(aptr + kk + 16);
    #pragma unroll
    for (int j = 0; j < 4; j++) {
      int e = e0 + j * 16 + lc;
      const _Float16* bp = W16 + (size_t)e * D_DIM + kk + (half ? 8 : 0);
      U16h Bm; Bm.h[0] = *(const v8h*)bp; Bm.h[1] = *(const v8h*)(bp + 16);
      acc[j] = __builtin_amdgcn_wmma_f32_16x16x32_f16(false, A.v, false, Bm.v,
                                                      (short)0, acc[j], false, false);
    }
  }
  gemm1_store(acc, b_in, Vpt, m0, e0, lc, half);
}

// ---------------- Kernel D (fallback path): GEMM1 from f32 V ----------------
__global__ void __launch_bounds__(256) k_gemm1_f32(const float* __restrict__ V,
                                                   const _Float16* __restrict__ W16,
                                                   const float* __restrict__ b_in,
                                                   _Float16* __restrict__ Vpt) {
  int wid  = threadIdx.x >> 5;
  int lane = threadIdx.x & 31;
  int gw   = blockIdx.x * 8 + wid;
  int mt   = gw / 12, ng = gw % 12;
  int m0   = mt * 16, e0 = ng * 64;
  int half = (lane < 16) ? 0 : 1;
  int lc   = lane & 15;

  v8f acc[4] = {};
  const float* aptr = V + (size_t)(m0 + lc) * D_DIM + (half ? 8 : 0);

  for (int kk = 0; kk < D_DIM; kk += 32) {
    U16h A;
    v8f f0 = *(const v8f*)(aptr + kk);
    v8f f1 = *(const v8f*)(aptr + kk + 16);
    #pragma unroll
    for (int i = 0; i < 8; i++) { A.h[0][i] = (_Float16)f0[i]; A.h[1][i] = (_Float16)f1[i]; }
    #pragma unroll
    for (int j = 0; j < 4; j++) {
      int e = e0 + j * 16 + lc;
      const _Float16* bp = W16 + (size_t)e * D_DIM + kk + (half ? 8 : 0);
      U16h Bm; Bm.h[0] = *(const v8h*)bp; Bm.h[1] = *(const v8h*)(bp + 16);
      acc[j] = __builtin_amdgcn_wmma_f32_16x16x32_f16(false, A.v, false, Bm.v,
                                                      (short)0, acc[j], false, false);
    }
  }
  gemm1_store(acc, b_in, Vpt, m0, e0, lc, half);
}

// ---------------- Kernel E: fused w -> pooled GEMM -> LN -> proj ------------
// one block per (b, 16 rows of T); 8 waves, each owns 6 d-tiles (96 cols)
__global__ void __launch_bounds__(256) k_fused(const float* __restrict__ tau,
                                               const float* __restrict__ t_hat,
                                               const float* __restrict__ mask,
                                               const float* __restrict__ log_sigma,
                                               const _Float16* __restrict__ Vpt,
                                               const float* __restrict__ gamma,
                                               const float* __restrict__ beta,
                                               const _Float16* __restrict__ Wp16,
                                               const float* __restrict__ b_proj,
                                               float* __restrict__ out) {
  __shared__ _Float16 w_lds[16 * N_NOTES];   // 16 KB : A-tile for pooling GEMM
  __shared__ _Float16 en_lds[16 * D_DIM];    // 24 KB : LayerNormed rows (f16)
  __shared__ float denom[16], rsum[16], rsq[16];

  int b  = blockIdx.x >> 6;                  // T/16 = 64 tiles per batch
  int t0 = (blockIdx.x & 63) * 16;
  int tid = threadIdx.x;
  int wid = tid >> 5, lane = tid & 31, lc = lane & 15, half = (lane < 16) ? 0 : 1;

  if (tid < 16) { denom[tid] = 0.f; rsum[tid] = 0.f; rsq[tid] = 0.f; }
  __syncthreads();

  // ---- phase 1: Gaussian recency weights into LDS + denom -----------------
  {
    float inv_s = 1.0f / __expf(log_sigma[0]);
    int m = tid & 15, chunk = tid >> 4;
    float th = t_hat[b * T_DIM + t0 + m];
    float lsum = 0.f;
    int nb = chunk * 32;
    const float4* tp = (const float4*)(tau  + b * N_NOTES + nb);
    const float4* mp = (const float4*)(mask + b * N_NOTES + nb);
    #pragma unroll
    for (int q = 0; q < 8; q++) {            // 8 * 4 = 32 notes
      float4 tv = tp[q];
      float4 mv = mp[q];
      float d0 = fmaxf(th - tv.x, 0.f) * inv_s;
      float d1 = fmaxf(th - tv.y, 0.f) * inv_s;
      float d2 = fmaxf(th - tv.z, 0.f) * inv_s;
      float d3 = fmaxf(th - tv.w, 0.f) * inv_s;
      float w0 = __expf(-d0 * d0) * mv.x;
      float w1 = __expf(-d1 * d1) * mv.y;
      float w2 = __expf(-d2 * d2) * mv.z;
      float w3 = __expf(-d3 * d3) * mv.w;
      v4h h; h[0] = (_Float16)w0; h[1] = (_Float16)w1;
             h[2] = (_Float16)w2; h[3] = (_Float16)w3;
      *(v4h*)&w_lds[m * N_NOTES + nb + q * 4] = h;
      lsum += (w0 + w1) + (w2 + w3);
    }
    atomicAdd(&denom[m], lsum);
  }
  __syncthreads();

  // ---- phase 2: E_wsum tile = w^T @ Vp  (K = N = 512) ---------------------
  v8f acc[6] = {};
  const _Float16* vb = Vpt + (size_t)b * D_DIM * N_NOTES;
  for (int n0 = 0; n0 < N_NOTES; n0 += 32) {
    U16h A;
    const _Float16* ap = &w_lds[lc * N_NOTES + n0 + (half ? 8 : 0)];
    A.h[0] = *(const v8h*)ap; A.h[1] = *(const v8h*)(ap + 16);
    #pragma unroll
    for (int j = 0; j < 6; j++) {
      int d = (wid * 6 + j) * 16 + lc;
      const _Float16* bp = vb + (size_t)d * N_NOTES + n0 + (half ? 8 : 0);
      U16h Bm; Bm.h[0] = *(const v8h*)bp; Bm.h[1] = *(const v8h*)(bp + 16);
      acc[j] = __builtin_amdgcn_wmma_f32_16x16x32_f16(false, A.v, false, Bm.v,
                                                      (short)0, acc[j], false, false);
    }
  }

  // ---- divide by denom; LayerNorm partial sums ----------------------------
  float rcp[8], ps[8], pq[8];
  #pragma unroll
  for (int r = 0; r < 8; r++) {
    rcp[r] = 1.0f / fmaxf(denom[r + (half ? 8 : 0)], 1e-6f);
    ps[r] = 0.f; pq[r] = 0.f;
  }
  #pragma unroll
  for (int j = 0; j < 6; j++)
    #pragma unroll
    for (int r = 0; r < 8; r++) {
      float v = acc[j][r] * rcp[r];
      acc[j][r] = v;
      ps[r] += v; pq[r] += v * v;
    }
  #pragma unroll
  for (int r = 0; r < 8; r++)
    #pragma unroll
    for (int off = 8; off; off >>= 1) {
      ps[r] += __shfl_xor(ps[r], off, 16);
      pq[r] += __shfl_xor(pq[r], off, 16);
    }
  if (lc == 0) {
    #pragma unroll
    for (int r = 0; r < 8; r++) {
      atomicAdd(&rsum[r + (half ? 8 : 0)], ps[r]);
      atomicAdd(&rsq[r + (half ? 8 : 0)], pq[r]);
    }
  }
  __syncthreads();

  // ---- normalize + gamma/beta, stash f16 rows in LDS ----------------------
  #pragma unroll
  for (int j = 0; j < 6; j++) {
    int d = (wid * 6 + j) * 16 + lc;
    float g = gamma[d], be = beta[d];
    #pragma unroll
    for (int r = 0; r < 8; r++) {
      int m = r + (half ? 8 : 0);
      float mu  = rsum[m] * (1.0f / 768.0f);
      float var = rsq[m]  * (1.0f / 768.0f) - mu * mu;
      float rs  = rsqrtf(var + 1e-5f);
      float v   = (acc[j][r] - mu) * rs * g + be;
      en_lds[m * D_DIM + d] = (_Float16)v;
    }
  }
  __syncthreads();

  // ---- phase 4: output projection  E_txt = E_norm @ W_proj^T + b_proj -----
  v8f acc2[6] = {};
  for (int k0 = 0; k0 < D_DIM; k0 += 32) {
    U16h A;
    const _Float16* ap = &en_lds[lc * D_DIM + k0 + (half ? 8 : 0)];
    A.h[0] = *(const v8h*)ap; A.h[1] = *(const v8h*)(ap + 16);
    #pragma unroll
    for (int j = 0; j < 6; j++) {
      int e = (wid * 6 + j) * 16 + lc;
      const _Float16* bp = Wp16 + (size_t)e * D_DIM + k0 + (half ? 8 : 0);
      U16h Bm; Bm.h[0] = *(const v8h*)bp; Bm.h[1] = *(const v8h*)(bp + 16);
      acc2[j] = __builtin_amdgcn_wmma_f32_16x16x32_f16(false, A.v, false, Bm.v,
                                                       (short)0, acc2[j], false, false);
    }
  }
  float* outb = out + (size_t)b * T_DIM * D_DIM + (size_t)t0 * D_DIM;
  #pragma unroll
  for (int j = 0; j < 6; j++) {
    int e = (wid * 6 + j) * 16 + lc;
    float bias = b_proj[e];
    #pragma unroll
    for (int r = 0; r < 8; r++) {
      int m = r + (half ? 8 : 0);
      outb[(size_t)m * D_DIM + e] = acc2[j][r] + bias;
    }
  }
}

// ---------------------------------------------------------------------------
extern "C" void kernel_launch(void* const* d_in, const int* in_sizes, int n_in,
                              void* d_out, int out_size, void* d_ws, size_t ws_size,
                              hipStream_t stream) {
  const float* V         = (const float*)d_in[0];
  const float* tau       = (const float*)d_in[1];
  const float* t_hat     = (const float*)d_in[2];
  const float* W_in      = (const float*)d_in[3];
  const float* b_in      = (const float*)d_in[4];
  const float* ln_gamma  = (const float*)d_in[5];
  const float* ln_beta   = (const float*)d_in[6];
  const float* W_proj    = (const float*)d_in[7];
  const float* b_proj    = (const float*)d_in[8];
  const float* log_sigma = (const float*)d_in[9];

  // workspace layout
  _Float16* Win16 = (_Float16*)d_ws;                         // 768*768 f16
  _Float16* Wp16  = Win16 + D_DIM * D_DIM;                   // 768*768 f16
  float*    mask  = (float*)(Wp16 + D_DIM * D_DIM);          // B*N f32
  _Float16* Vpt   = (_Float16*)(mask + B_DIM * N_NOTES);     // B*D*N f16 (transposed Vp)
  _Float16* Vh    = Vpt + (size_t)B_DIM * D_DIM * N_NOTES;   // B*N*D f16 (optional)

  const size_t base_bytes = (size_t)2 * D_DIM * D_DIM * 2 +
                            (size_t)B_DIM * N_NOTES * 4 +
                            (size_t)B_DIM * D_DIM * N_NOTES * 2;
  const size_t f16_bytes  = base_bytes + (size_t)B_DIM * N_NOTES * D_DIM * 2;

  float* out   = (float*)d_out;
  float* out_m = out + (size_t)B_DIM * T_DIM * D_DIM;        // M_txt tail (B floats)

  k_cvt_w<<<(D_DIM * D_DIM + 255) / 256, 256, 0, stream>>>(W_in, W_proj, Win16, Wp16);

  if (ws_size >= f16_bytes) {
    // prepass: convert V to f16 (fused with note-mask), then f16-only GEMM1
    k_prep<<<(B_DIM * N_NOTES) / 8, 256, 0, stream>>>(V, Vh, mask);
    k_mtxt<<<B_DIM, 256, 0, stream>>>(mask, out_m);
    k_gemm1_f16<<<(2048 * 12) / 8, 256, 0, stream>>>(Vh, Win16, b_in, Vpt);
  } else {
    k_mask<<<(B_DIM * N_NOTES) / 8, 256, 0, stream>>>(V, mask);
    k_mtxt<<<B_DIM, 256, 0, stream>>>(mask, out_m);
    k_gemm1_f32<<<(2048 * 12) / 8, 256, 0, stream>>>(V, Win16, b_in, Vpt);
  }

  k_fused<<<B_DIM * (T_DIM / 16), 256, 0, stream>>>(tau, t_hat, mask, log_sigma,
                                                    Vpt, ln_gamma, ln_beta, Wp16,
                                                    b_proj, out);
}
